// NeuralMetaLearner_73272142070136
// MI455X (gfx1250) — matrix operand order repaired and checked by reading, not compiled, not run on hardware
//
#include <hip/hip_runtime.h>
#include <cmath>

typedef __attribute__((ext_vector_type(16))) _Float16 v16h;
typedef __attribute__((ext_vector_type(8)))  _Float16 v8h;
typedef __attribute__((ext_vector_type(8)))  float    v8f;

#define NWAVES   8      // 256 threads = 8 wave32
#define NBLOCKS  2048

// Build a 16x32 f16 A-fragment from fp32 memory (global or LDS-backed generic ptr).
// Lane L (m = L&15, hi = L>>4): halves 0..7  = K(k0 + 8*hi + 0..7)
//                               halves 8..15 = K(k0 + 16 + 8*hi + 0..7)
// Caller passes p = row_base + k0 + 8*hi.
static __device__ inline v16h load_a_frag_f32(const float* p) {
    float4 f0 = *(const float4*)(p);
    float4 f1 = *(const float4*)(p + 4);
    float4 f2 = *(const float4*)(p + 16);
    float4 f3 = *(const float4*)(p + 20);
    v16h a;
    a[0]=(_Float16)f0.x;  a[1]=(_Float16)f0.y;  a[2]=(_Float16)f0.z;  a[3]=(_Float16)f0.w;
    a[4]=(_Float16)f1.x;  a[5]=(_Float16)f1.y;  a[6]=(_Float16)f1.z;  a[7]=(_Float16)f1.w;
    a[8]=(_Float16)f2.x;  a[9]=(_Float16)f2.y;  a[10]=(_Float16)f2.z; a[11]=(_Float16)f2.w;
    a[12]=(_Float16)f3.x; a[13]=(_Float16)f3.y; a[14]=(_Float16)f3.z; a[15]=(_Float16)f3.w;
    return a;
}

// 32x16 f16 B-fragment: lane L (n = n0 + (L&15), hi = L>>4) holds halves
// W[n, k0 + 8*hi + 0..7] then W[n, k0 + 16 + 8*hi + 0..7].
static __device__ inline v16h load_b_frag(const _Float16* p) {
    v8h lo = *(const v8h*)(p);
    v8h hi8 = *(const v8h*)(p + 16);
    return __builtin_shufflevector(lo, hi8, 0,1,2,3,4,5,6,7,8,9,10,11,12,13,14,15);
}

// Exact-erf GELU via Abramowitz-Stegun 7.1.26 (|eps| <= 1.5e-7 in erf):
// ~12 VALU (v_rcp + 6 fma + v_exp + abs/sign) vs ~25+ for libm erff.
static __device__ inline float gelu_fast(float v) {
    float x  = v * 0.70710678118654752f;       // v / sqrt(2)
    float ax = __builtin_fabsf(x);
    float t  = __builtin_amdgcn_rcpf(__builtin_fmaf(0.3275911f, ax, 1.0f));
    float p  = __builtin_fmaf(1.061405429f, t, -1.453152027f);
    p = __builtin_fmaf(p, t,  1.421413741f);
    p = __builtin_fmaf(p, t, -0.284496736f);
    p = __builtin_fmaf(p, t,  0.254829592f);
    p = p * t;
    float e  = __expf(-ax * ax);
    float er = __builtin_fmaf(-p, e, 1.0f);    // erf(|x|)
    float s  = __builtin_copysignf(er, x);     // erf(x)
    return 0.5f * v * (1.0f + s);
}

__global__ __launch_bounds__(256) void meta_learner_fused(
    const float* __restrict__ x, const int* __restrict__ regime,
    const float* __restrict__ W1, const float* __restrict__ b1,
    const float* __restrict__ gamma, const float* __restrict__ beta,
    const float* __restrict__ W2, const float* __restrict__ b2,
    const float* __restrict__ Wg, const float* __restrict__ bg,
    const float* __restrict__ Wr, const float* __restrict__ br,
    const float* __restrict__ emb, const float* __restrict__ log_temp,
    float* __restrict__ out, int B)
{
    __shared__ _Float16 w1h[64 * 128];          // 16 KB, f16 row-major
    __shared__ _Float16 w2h[32 * 64];           //  4 KB, f16 row-major
    __shared__ float    weffL[2][4][32];        // folded head weights: 0.3*Wr + 0.2*Wg*sigmoid(emb)
    __shared__ float    beffL[4][2];            // folded head bias:    0.7*bg + 0.3*br
    __shared__ float    g1s[NWAVES][16 * 68];   // per-wave trunk scratch (stride 68: 16B-aligned, conflict-free)
    __shared__ float    h2s[NWAVES][16 * 36];   // per-wave h2 scratch   (stride 36: 16B-aligned, conflict-free)

    const int tid = threadIdx.x;
    for (int i = tid; i < 64 * 128; i += 256) w1h[i] = (_Float16)W1[i];
    for (int i = tid; i < 32 * 64;  i += 256) w2h[i] = (_Float16)W2[i];
    {   // one element per thread: 2 outputs x 4 regimes x 32 cols == 256
        const int o = tid >> 7, r = (tid >> 5) & 3, c = tid & 31;
        float sig = __builtin_amdgcn_rcpf(1.0f + __expf(-emb[r * 32 + c]));
        weffL[o][r][c] = 0.3f * Wr[(size_t)(r * 2 + o) * 32 + c] + 0.2f * Wg[o * 32 + c] * sig;
        if (tid < 8) beffL[tid >> 1][tid & 1] = 0.7f * bg[tid & 1] + 0.3f * br[(tid >> 1) * 2 + (tid & 1)];
    }
    __syncthreads();

    const int lane = tid & 31;
    const int wid  = tid >> 5;
    const int li   = lane & 15;
    const int hi   = lane >> 4;

    // Per-lane constant preloads (tiny, L2-resident)
    float b1v[4], gamv[4], betv[4];
#pragma unroll
    for (int t = 0; t < 4; ++t) {
        b1v[t]  = b1[t * 16 + li];
        gamv[t] = gamma[t * 16 + li];
        betv[t] = beta[t * 16 + li];
    }
    float b2v[2] = { b2[li], b2[16 + li] };
    float wg0r[16], wg1r[16];
#pragma unroll
    for (int j = 0; j < 16; ++j) {
        wg0r[j] = Wg[hi * 16 + j];
        wg1r[j] = Wg[32 + hi * 16 + j];
    }
    float tempv = __expf(log_temp[0]);
    tempv = fminf(fmaxf(tempv, 0.5f), 5.0f);
    const float invt = 1.0f / tempv;

    const int nTiles  = B >> 4;
    const int tStride = (int)gridDim.x * NWAVES;

    for (int t = (int)blockIdx.x * NWAVES + wid; t < nTiles; t += tStride) {
        const int m0 = t << 4;

        // ---------------- GEMM1: h[16,64] = x[16,128] @ W1^T, f16 WMMA, f32 acc
        v8f acc[4];
#pragma unroll
        for (int tn = 0; tn < 4; ++tn) acc[tn] = v8f{0,0,0,0,0,0,0,0};

        const float* xrow = x + (size_t)(m0 + li) * 128 + hi * 8;
        if (t + tStride < nTiles) {
            __builtin_prefetch(xrow + (size_t)tStride * 16 * 128, 0, 0);  // global_prefetch_b8
            __builtin_prefetch(xrow + (size_t)tStride * 16 * 128 + 64, 0, 0);
        }

#pragma unroll
        for (int ks = 0; ks < 4; ++ks) {
            const v16h a = load_a_frag_f32(xrow + ks * 32);
#pragma unroll
            for (int tn = 0; tn < 4; ++tn) {
                const v16h bf = load_b_frag(&w1h[(tn * 16 + li) * 128 + ks * 32 + hi * 8]);
                acc[tn] = __builtin_amdgcn_wmma_f32_16x16x32_f16(
                    false, a, false, bf, (short)0, acc[tn], false, false);
            }
        }

        // ---------------- bias + LayerNorm (reduce across 16-lane halves)
        float mu[8], rsd[8];
#pragma unroll
        for (int r = 0; r < 8; ++r) {
            float v0 = acc[0][r] + b1v[0];
            float v1 = acc[1][r] + b1v[1];
            float v2 = acc[2][r] + b1v[2];
            float v3 = acc[3][r] + b1v[3];
            acc[0][r] = v0; acc[1][r] = v1; acc[2][r] = v2; acc[3][r] = v3;
            float s = v0 + v1 + v2 + v3;
            float q = v0 * v0 + v1 * v1 + v2 * v2 + v3 * v3;
#pragma unroll
            for (int msk = 1; msk < 16; msk <<= 1) {
                s += __shfl_xor(s, msk, 32);
                q += __shfl_xor(q, msk, 32);
            }
            float m = s * (1.0f / 64.0f);
            mu[r]  = m;
            rsd[r] = rsqrtf(q * (1.0f / 64.0f) - m * m + 1e-5f);
        }

        // ---------------- affine + GELU, transpose via per-wave LDS
        float* gbase = &g1s[wid][0];
#pragma unroll
        for (int tn = 0; tn < 4; ++tn) {
#pragma unroll
            for (int r = 0; r < 8; ++r) {
                float v = (acc[tn][r] - mu[r]) * rsd[r] * gamv[tn] + betv[tn];
                gbase[(r + 8 * hi) * 68 + tn * 16 + li] = gelu_fast(v);
            }
        }

        // ---------------- GEMM2: h2[16,32] = g1[16,64] @ W2^T
        v8f acc2[2] = { v8f{0,0,0,0,0,0,0,0}, v8f{0,0,0,0,0,0,0,0} };
#pragma unroll
        for (int ks = 0; ks < 2; ++ks) {
            const v16h a = load_a_frag_f32(&gbase[li * 68 + ks * 32 + hi * 8]);
#pragma unroll
            for (int tn = 0; tn < 2; ++tn) {
                const v16h bf = load_b_frag(&w2h[(tn * 16 + li) * 64 + ks * 32 + hi * 8]);
                acc2[tn] = __builtin_amdgcn_wmma_f32_16x16x32_f16(
                    false, a, false, bf, (short)0, acc2[tn], false, false);
            }
        }

        // ---------------- bias + GELU, stage h2 for the heads
        float* hbase = &h2s[wid][0];
#pragma unroll
        for (int tn = 0; tn < 2; ++tn) {
#pragma unroll
            for (int r = 0; r < 8; ++r) {
                float v = gelu_fast(acc2[tn][r] + b2v[tn]);
                hbase[(r + 8 * hi) * 36 + tn * 16 + li] = v;
            }
        }

        // ---------------- heads: lane owns row li, half hi (16 of 32 cols)
        const float* hp = &hbase[li * 36 + hi * 16];
        float hb[16];
#pragma unroll
        for (int j = 0; j < 16; ++j) hb[j] = hp[j];

        const int rg = regime[m0 + li];
        const float* we0 = &weffL[0][rg][hi * 16];
        const float* we1 = &weffL[1][rg][hi * 16];

        float g0 = 0.f, g1 = 0.f, e0 = 0.f, e1 = 0.f;
#pragma unroll
        for (int j = 0; j < 16; ++j) {
            float hv = hb[j];
            g0 += hv * wg0r[j];
            g1 += hv * wg1r[j];
            e0 += hv * we0[j];
            e1 += hv * we1[j];
        }
        g0 += __shfl_xor(g0, 16, 32);  g1 += __shfl_xor(g1, 16, 32);
        e0 += __shfl_xor(e0, 16, 32);  e1 += __shfl_xor(e1, 16, 32);

        if (hi == 0) {
            float l0 = 0.5f * g0 + e0 + beffL[rg][0];
            float l1 = 0.5f * g1 + e1 + beffL[rg][1];
            float2 o;
            o.x = l0 * invt;
            o.y = l1 * invt;
            *(float2*)(out + (size_t)(m0 + li) * 2) = o;
        }
    }
}

extern "C" void kernel_launch(void* const* d_in, const int* in_sizes, int n_in,
                              void* d_out, int out_size, void* d_ws, size_t ws_size,
                              hipStream_t stream) {
    const float* x        = (const float*)d_in[0];
    const int*   regime   = (const int*)  d_in[1];
    const float* W1       = (const float*)d_in[2];
    const float* b1       = (const float*)d_in[3];
    const float* gamma    = (const float*)d_in[4];
    const float* beta     = (const float*)d_in[5];
    const float* W2       = (const float*)d_in[6];
    const float* b2       = (const float*)d_in[7];
    const float* Wg       = (const float*)d_in[8];
    const float* bg       = (const float*)d_in[9];
    const float* Wr       = (const float*)d_in[10];
    const float* br       = (const float*)d_in[11];
    const float* emb      = (const float*)d_in[12];
    const float* log_temp = (const float*)d_in[13];
    float* out = (float*)d_out;

    const int B      = in_sizes[0] / 128;
    const int nTiles = B >> 4;
    int blocks = NBLOCKS;
    if (blocks * NWAVES > nTiles) blocks = (nTiles + NWAVES - 1) / NWAVES;
    if (blocks < 1) blocks = 1;

    meta_learner_fused<<<blocks, 256, 0, stream>>>(
        x, regime, W1, b1, gamma, beta, W2, b2, Wg, bg, Wr, br, emb, log_temp,
        out, B);
}
